// FP_GNN_11845519803008
// MI455X (gfx1250) — compile-verified, matrix operand order checked by reading, not run on hardware
//
#include <hip/hip_runtime.h>
#include <hip/hip_bf16.h>
#include <math.h>

// ---------------------------------------------------------------------------
// FP_GNN on MI455X (gfx1250, wave32).  bf16 WMMA for both GEMM stages;
// attention matmul shares B-tiles across 4 waves via async global->LDS
// (ASYNCcnt) double buffering.
//   N=4096, F_IN=68 (padded to 96), HID=64, G=128, FP_DIM=167, LAYERS=3
// ---------------------------------------------------------------------------

typedef __bf16 bf16_t;
typedef __attribute__((ext_vector_type(16))) __bf16 v16bf;
typedef __attribute__((ext_vector_type(8)))  __bf16 v8bf;
typedef __attribute__((ext_vector_type(8)))  float  v8f;

#define NN   4096
#define HID  64
#define GG   128
#define FPD  167
#define K0P  96      // 68 padded up to multiple of 32
#define LDSPAD 40    // halves per staged column (80B stride: breaks bank conflicts)

static __device__ __forceinline__ bf16_t f2bf(float f) {
  union { float f; unsigned u; } in; in.f = f;
  unsigned u = in.u;
  u += 0x7fffu + ((u >> 16) & 1u);            // round-to-nearest-even
  unsigned short h = (unsigned short)(u >> 16);
  union { unsigned short s; bf16_t b; } out; out.s = h;
  return out.b;
}

// ---------------------------- prep kernels ---------------------------------

__global__ void prep_x(const float* __restrict__ x, bf16_t* __restrict__ xpad) {
  int idx = blockIdx.x * blockDim.x + threadIdx.x;
  if (idx >= NN * K0P) return;
  int row = idx / K0P, k = idx % K0P;
  xpad[idx] = (k < 68) ? f2bf(x[row * 68 + k]) : f2bf(0.f);
}

__global__ void prep_w0(const float* __restrict__ W0, bf16_t* __restrict__ W0T) {
  int idx = blockIdx.x * blockDim.x + threadIdx.x;
  if (idx >= HID * K0P) return;
  int c = idx / K0P, k = idx % K0P;                       // W0T[c][k] = W0[k][c]
  W0T[idx] = (k < 68) ? f2bf(W0[k * HID + c]) : f2bf(0.f);
}

// tile(out,4)@W == out@(sum of 4 row-blocks of W); store transposed bf16
__global__ void prep_weff(const float* __restrict__ W1, const float* __restrict__ W2,
                          bf16_t* __restrict__ W1T, bf16_t* __restrict__ W2T) {
  int idx = blockIdx.x * blockDim.x + threadIdx.x;
  if (idx >= 2 * HID * HID) return;
  const float* W  = (idx < HID * HID) ? W1 : W2;
  bf16_t*      WT = (idx < HID * HID) ? W1T : W2T;
  int rem = idx & (HID * HID - 1);
  int c = rem >> 6, k = rem & 63;                         // WT[c][k] = sum_b W[b*64+k][c]
  float acc = 0.f;
  #pragma unroll
  for (int b = 0; b < 4; ++b) acc += W[(size_t)(b * HID + k) * HID + c];
  WT[rem] = f2bf(acc);
}

// ------------------- WMMA GEMM:  C[N x 64] = A[N x K] * BT^T ----------------
// A row-major bf16, BT[64][K] bf16 (B transposed), K multiple of 32.
// One wave per 16 output rows.  A-fragment lane layout per CDNA5 ISA 7.12.2:
// lane = M row (mod 16); K chunks [8*half,+8) and [16+8*half,+8).

__global__ __launch_bounds__(32)
void gemm_wmma(const bf16_t* __restrict__ A, int K,
               const bf16_t* __restrict__ BT, float* __restrict__ C) {
  const int lane = threadIdx.x;
  const int r = lane & 15, half = lane >> 4;
  const int rowBase = blockIdx.x * 16;
  const int row = rowBase + r;

  v8f acc[4] = {};
  for (int kk = 0; kk < K; kk += 32) {
    v16bf a;
    {
      const bf16_t* ap = A + (size_t)row * K + kk + 8 * half;
      v8bf lo = *(const v8bf*)(ap);
      v8bf hi = *(const v8bf*)(ap + 16);
      #pragma unroll
      for (int t = 0; t < 8; ++t) { a[t] = lo[t]; a[8 + t] = hi[t]; }
    }
    #pragma unroll
    for (int nt = 0; nt < 4; ++nt) {
      const bf16_t* bp = BT + (size_t)(nt * 16 + r) * K + kk + 8 * half;
      v8bf lo = *(const v8bf*)(bp);
      v8bf hi = *(const v8bf*)(bp + 16);
      v16bf b;
      #pragma unroll
      for (int t = 0; t < 8; ++t) { b[t] = lo[t]; b[8 + t] = hi[t]; }
      acc[nt] = __builtin_amdgcn_wmma_f32_16x16x32_bf16(
          false, a, false, b, (short)0, acc[nt], false, false);
    }
  }
  #pragma unroll
  for (int nt = 0; nt < 4; ++nt)
    #pragma unroll
    for (int rr = 0; rr < 8; ++rr)
      C[(size_t)(rowBase + rr + 8 * half) * HID + nt * 16 + r] = acc[nt][rr];
}

// ------------------------- per-row s, d projections ------------------------

__global__ void sd_kernel(const float* __restrict__ Wh,
                          const float* __restrict__ asrc, const float* __restrict__ adst,
                          float* __restrict__ s, float* __restrict__ d) {
  int i = blockIdx.x * blockDim.x + threadIdx.x;
  if (i >= NN) return;
  const float* w = Wh + (size_t)i * HID;
  float ss = 0.f, dd = 0.f;
  #pragma unroll 8
  for (int k = 0; k < HID; ++k) { ss += w[k] * asrc[k]; dd += w[k] * adst[k]; }
  s[i] = ss; d[i] = dd;
}

// ------------------------ Wh -> bf16 transpose (B operand) -----------------

__global__ void conv_whT(const float* __restrict__ Wh, bf16_t* __restrict__ WhT) {
  int idx = blockIdx.x * blockDim.x + threadIdx.x;
  if (idx >= NN * HID) return;
  int row = idx >> 6, col = idx & 63;
  WhT[(size_t)col * NN + row] = f2bf(Wh[idx]);
}

// --------------- online-softmax row stats (8 rows per 256-thr block) -------

__global__ __launch_bounds__(256)
void rowstats(const float* __restrict__ adj, const float* __restrict__ s,
              const float* __restrict__ d, float* __restrict__ m_out,
              float* __restrict__ rz_out) {
  const int lane = threadIdx.x & 31;
  const int row  = blockIdx.x * 8 + (threadIdx.x >> 5);
  const float sR = s[row];
  const float* arow = adj + (size_t)row * NN;
  float m = -1e30f, l = 0.f;
  for (int j = lane; j < NN; j += 32) {
    if (arow[j] > 0.f) {
      float v = sR + d[j];
      v = v > 0.f ? v : 0.2f * v;                    // leaky_relu(0.2)
      float nm = fmaxf(m, v);
      l = l * __expf(m - nm) + __expf(v - nm);
      m = nm;
    }
  }
  #pragma unroll
  for (int off = 16; off >= 1; off >>= 1) {
    float mo = __shfl_xor(m, off, 32);
    float lo = __shfl_xor(l, off, 32);
    float nm = fmaxf(m, mo);
    l = l * __expf(m - nm) + lo * __expf(mo - nm);
    m = nm;
  }
  if (lane == 0) { m_out[row] = m; rz_out[row] = 1.0f / l; }
}

// ------- fused masked-softmax P tile (on the fly, bf16) x Wh via WMMA ------
// 4 waves / block, each owning a 16-row tile; the 4KB B chunk Wh^T[0:64][j:j+32]
// is staged once per block into LDS with async global->LDS loads (ASYNCcnt),
// double-buffered, then read as WMMA B fragments with ds_load_b128.

__global__ __launch_bounds__(128)
void attn_wmma(const float* __restrict__ adj, const float* __restrict__ s,
               const float* __restrict__ d, const float* __restrict__ m,
               const float* __restrict__ rz, const bf16_t* __restrict__ WhT,
               float* __restrict__ outF, bf16_t* __restrict__ hb) {
  __shared__ __align__(16) bf16_t bbuf[2][HID * LDSPAD];

  const int tid  = threadIdx.x;
  const int lane = tid & 31;
  const int wave = tid >> 5;                    // 0..3
  const int r = lane & 15, half = lane >> 4;
  const int rowBase = (blockIdx.x * 4 + wave) * 16;
  const int row = rowBase + r;
  const float sR = s[row], mR = m[row], zR = rz[row];
  const float* arow = adj + (size_t)row * NN;

  // staging: 64 cols x 4 segs (8 halves each) = 256 b128 items, 128 threads x 2
  const int it0 = tid, it1 = tid + 128;
  const int col0 = it0 >> 2, seg0 = it0 & 3;
  const int col1 = it1 >> 2, seg1 = it1 & 3;
  const bf16_t* g0 = WhT + (size_t)col0 * NN + seg0 * 8;
  const bf16_t* g1 = WhT + (size_t)col1 * NN + seg1 * 8;
  const unsigned l0[2] = {
    (unsigned)(unsigned long long)(const void*)&bbuf[0][col0 * LDSPAD + seg0 * 8],
    (unsigned)(unsigned long long)(const void*)&bbuf[1][col0 * LDSPAD + seg0 * 8]};
  const unsigned l1[2] = {
    (unsigned)(unsigned long long)(const void*)&bbuf[0][col1 * LDSPAD + seg1 * 8],
    (unsigned)(unsigned long long)(const void*)&bbuf[1][col1 * LDSPAD + seg1 * 8]};

  #define ISSUE_CHUNK(BUF, J)                                                  \
    do {                                                                       \
      unsigned long long ga0 = (unsigned long long)(const void*)(g0 + (J));    \
      unsigned long long ga1 = (unsigned long long)(const void*)(g1 + (J));    \
      asm volatile("global_load_async_to_lds_b128 %0, %1, off"                 \
                   :: "v"(l0[BUF]), "v"(ga0) : "memory");                      \
      asm volatile("global_load_async_to_lds_b128 %0, %1, off"                 \
                   :: "v"(l1[BUF]), "v"(ga1) : "memory");                      \
    } while (0)

  ISSUE_CHUNK(0, 0);

  v8f acc[4] = {};
  for (int i = 0; i < NN / 32; ++i) {
    const int j = i * 32;
    const int cur = i & 1;
    if (i + 1 < NN / 32) {
      ISSUE_CHUNK(cur ^ 1, j + 32);
      asm volatile("s_wait_asynccnt 0x2" ::: "memory");  // chunk i complete
    } else {
      asm volatile("s_wait_asynccnt 0x0" ::: "memory");
    }
    __syncthreads();                                     // buf[cur] visible to all

    // ---- A fragment: softmaxed attention row slice, built on the fly ----
    const int c0 = j + 8 * half;
    v16bf a;
    #pragma unroll
    for (int t = 0; t < 8; ++t) {
      int col = c0 + t;
      float v = sR + d[col];
      v = v > 0.f ? v : 0.2f * v;
      float arg = (arow[col] > 0.f) ? (v - mR) : -1e30f;   // exp(-1e30)==0
      a[t] = f2bf(__expf(arg) * zR);
    }
    #pragma unroll
    for (int t = 0; t < 8; ++t) {
      int col = c0 + 16 + t;
      float v = sR + d[col];
      v = v > 0.f ? v : 0.2f * v;
      float arg = (arow[col] > 0.f) ? (v - mR) : -1e30f;
      a[8 + t] = f2bf(__expf(arg) * zR);
    }

    // ---- B fragments from LDS, 4 output column tiles ----
    #pragma unroll
    for (int nt = 0; nt < 4; ++nt) {
      const bf16_t* bp = &bbuf[cur][(nt * 16 + r) * LDSPAD + 8 * half];
      v8bf lo = *(const v8bf*)(bp);
      v8bf hi = *(const v8bf*)(bp + 16);
      v16bf b;
      #pragma unroll
      for (int t = 0; t < 8; ++t) { b[t] = lo[t]; b[8 + t] = hi[t]; }
      acc[nt] = __builtin_amdgcn_wmma_f32_16x16x32_bf16(
          false, a, false, b, (short)0, acc[nt], false, false);
    }
    __syncthreads();                                     // reads done before reuse
  }
  #undef ISSUE_CHUNK

  #pragma unroll
  for (int nt = 0; nt < 4; ++nt)
    #pragma unroll
    for (int rr = 0; rr < 8; ++rr) {
      float v = fmaxf(acc[nt][rr], 0.f);                 // relu (both branches)
      size_t o = (size_t)(rowBase + rr + 8 * half) * HID + nt * 16 + r;
      outF[o] = v;
      hb[o]   = f2bf(v);
    }
}

// ---------------- deterministic segment-mean pool (batch sorted) -----------

__global__ void pool_kernel(const float* __restrict__ h, const int* __restrict__ batch,
                            float* __restrict__ gnn) {
  int idx = blockIdx.x * blockDim.x + threadIdx.x;       // G*HID
  if (idx >= GG * HID) return;
  int g = idx >> 6, f = idx & 63;
  int lo = 0, hi = NN;                                   // first index with batch >= g
  while (lo < hi) { int mid = (lo + hi) >> 1; if (batch[mid] < g) lo = mid + 1; else hi = mid; }
  int start = lo;
  lo = start; hi = NN;                                   // first index with batch > g
  while (lo < hi) { int mid = (lo + hi) >> 1; if (batch[mid] <= g) lo = mid + 1; else hi = mid; }
  int end = lo;
  float acc = 0.f;
  for (int n = start; n < end; ++n) acc += h[(size_t)n * HID + f];
  gnn[idx] = acc / fmaxf((float)(end - start), 1.f);
}

// ------------------- fingerprint branch + final linear ---------------------

__global__ void final_kernel(const float* __restrict__ gnn, const float* __restrict__ fp,
                             const float* __restrict__ fc1_w, const float* __restrict__ fc1_b,
                             const float* __restrict__ fc_w, const float* __restrict__ fc_b,
                             float* __restrict__ out) {
  int g = threadIdx.x;
  if (g >= GG) return;
  float acc = fc_b[0];
  for (int f = 0; f < HID; ++f) {
    float fe = fc1_b[f];
    for (int k = 0; k < FPD; ++k) fe += fp[g * FPD + k] * fc1_w[k * HID + f];
    fe = fmaxf(fe, 0.f);
    acc += gnn[g * HID + f] * fc_w[f] + fe * fc_w[HID + f];
  }
  out[g] = fmaxf(acc, 0.f);
}

// ---------------------------------------------------------------------------

extern "C" void kernel_launch(void* const* d_in, const int* in_sizes, int n_in,
                              void* d_out, int out_size, void* d_ws, size_t ws_size,
                              hipStream_t stream) {
  const float* x     = (const float*)d_in[0];
  const float* adj   = (const float*)d_in[1];
  const int*   batch = (const int*)d_in[2];
  const float* fp    = (const float*)d_in[3];
  const float* W[3]    = {(const float*)d_in[4],  (const float*)d_in[7],  (const float*)d_in[10]};
  const float* asrc[3] = {(const float*)d_in[5],  (const float*)d_in[8],  (const float*)d_in[11]};
  const float* adst[3] = {(const float*)d_in[6],  (const float*)d_in[9],  (const float*)d_in[12]};
  const float* fc1_w = (const float*)d_in[13];
  const float* fc1_b = (const float*)d_in[14];
  const float* fc_w  = (const float*)d_in[15];
  const float* fc_b  = (const float*)d_in[16];
  float* out = (float*)d_out;

  // workspace carve-out (~4.06 MB total)
  char* p = (char*)d_ws;
  auto take = [&](size_t bytes) -> char* {
    char* r = p; p += (bytes + 255) & ~(size_t)255; return r;
  };
  bf16_t* xpad = (bf16_t*)take((size_t)NN * K0P * 2);
  bf16_t* w0t  = (bf16_t*)take((size_t)HID * K0P * 2);
  bf16_t* w1t  = (bf16_t*)take((size_t)HID * HID * 2);
  bf16_t* w2t  = (bf16_t*)take((size_t)HID * HID * 2);
  float*  Wh   = (float*) take((size_t)NN * HID * 4);
  bf16_t* WhT  = (bf16_t*)take((size_t)NN * HID * 2);
  bf16_t* hb   = (bf16_t*)take((size_t)NN * HID * 2);
  float*  outF = (float*) take((size_t)NN * HID * 4);
  float*  sb   = (float*) take((size_t)NN * 4);
  float*  db   = (float*) take((size_t)NN * 4);
  float*  mb   = (float*) take((size_t)NN * 4);
  float*  zb   = (float*) take((size_t)NN * 4);
  float*  gnn  = (float*) take((size_t)GG * HID * 4);

  prep_x   <<<(NN * K0P + 255) / 256, 256, 0, stream>>>(x, xpad);
  prep_w0  <<<(HID * K0P + 255) / 256, 256, 0, stream>>>(W[0], w0t);
  prep_weff<<<(2 * HID * HID + 255) / 256, 256, 0, stream>>>(W[1], W[2], w1t, w2t);

  for (int layer = 0; layer < 3; ++layer) {
    const bf16_t* A  = (layer == 0) ? xpad : hb;
    const int     K  = (layer == 0) ? K0P : HID;
    const bf16_t* BT = (layer == 0) ? w0t : (layer == 1 ? w1t : w2t);

    gemm_wmma<<<NN / 16, 32, 0, stream>>>(A, K, BT, Wh);
    sd_kernel<<<NN / 128, 128, 0, stream>>>(Wh, asrc[layer], adst[layer], sb, db);
    conv_whT <<<(NN * HID) / 256, 256, 0, stream>>>(Wh, WhT);
    rowstats <<<NN / 8, 256, 0, stream>>>(adj, sb, db, mb, zb);
    attn_wmma<<<NN / 64, 128, 0, stream>>>(adj, sb, db, mb, zb, WhT, outF, hb);
  }

  pool_kernel <<<(GG * HID + 255) / 256, 256, 0, stream>>>(outF, batch, gnn);
  final_kernel<<<1, 128, 0, stream>>>(gnn, fp, fc1_w, fc1_b, fc_w, fc_b, out);
  (void)in_sizes; (void)n_in; (void)out_size; (void)ws_size;
}
